// LoraSequential_59459527246489
// MI455X (gfx1250) — matrix operand order repaired and checked by reading, not compiled, not run on hardware
//
#include <hip/hip_runtime.h>

typedef __attribute__((ext_vector_type(16))) _Float16 v16h;
typedef __attribute__((ext_vector_type(8)))  _Float16 v8h;
typedef __attribute__((ext_vector_type(2)))  _Float16 h2;
typedef __attribute__((ext_vector_type(8)))  float    v8f;
typedef __attribute__((ext_vector_type(4)))  unsigned int u32x4;

union Frag   { v16h v; v8h h[2]; };
union Pack16 { u32x4 u; _Float16 f[8]; };

#define FDIM 4096
#define RANK 16
#define SEG  512
#define BM   128
#define BN   128
#define BK   32
#define LDA  40   // padded LDS stride (halves); 80B rows, 16B-aligned

static __device__ __forceinline__ v8f wmma_f16(v16h a, v16h b, v8f c) {
    return __builtin_amdgcn_wmma_f32_16x16x32_f16(false, a, false, b, (short)0, c, false, false);
}

// ------------- Kernel 0: AgT[g][r][k] = loraA[g][k][r]  (1MB, one-shot) ----
__global__ __launch_bounds__(256) void transposeA_kernel(
    const _Float16* __restrict__ loraA, _Float16* __restrict__ AgT) {
  __shared__ _Float16 s[256 * 17];
  const int t  = threadIdx.x;
  const int g  = blockIdx.x >> 4;
  const int k0 = (blockIdx.x & 15) << 8;          // 256 k-rows per block
  const _Float16* src = loraA + ((size_t)g * FDIM + k0 + t) * RANK;
  v8h r0 = *(const v8h*)src;
  v8h r1 = *(const v8h*)(src + 8);
#pragma unroll
  for (int i = 0; i < 8; ++i) { s[t * 17 + i] = r0[i]; s[t * 17 + 8 + i] = r1[i]; }
  __syncthreads();
  const int r  = t >> 4;                          // rank 0..15
  const int kc = (t & 15) << 4;                   // k chunk of 16
  _Float16 tmp[16];
#pragma unroll
  for (int i = 0; i < 16; ++i) tmp[i] = s[(kc + i) * 17 + r];
  _Float16* dst = AgT + ((size_t)g * RANK + r) * FDIM + k0 + kc;
#pragma unroll
  for (int i = 0; i < 16; ++i) dst[i] = tmp[i];
}

// ------------- Kernel 1: h[4096][16] = 2 * (x_g @ A_g) ---------------------
// 256 blocks x 128 threads; 4 waves split K, reduce partials through LDS.
// B fragments now read from AgT: two contiguous b128 loads per lane.
__global__ __launch_bounds__(128) void lora_h_kernel(
    const _Float16* __restrict__ x, const _Float16* __restrict__ AgT,
    _Float16* __restrict__ h) {
  __shared__ float red[4][8 * 32];
  const int wave = threadIdx.x >> 5;
  const int lane = threadIdx.x & 31;
  const int lm   = lane & 15;
  const int lk8  = (lane >> 4) << 3;
  const int lk16 = (lane >> 4) << 4;
  const int row0 = blockIdx.x * 16;
  const _Float16* Bt = AgT + ((size_t)(row0 / SEG) * RANK + lm) * FDIM;

  v8f acc = {0.f, 0.f, 0.f, 0.f, 0.f, 0.f, 0.f, 0.f};
  const int kbase = wave * (FDIM / 4);
  for (int kt = 0; kt < (FDIM / 4) / BK; ++kt) {
    const int kb = kbase + kt * BK;
    Frag a, b;
    const _Float16* xr = x + (size_t)(row0 + lm) * FDIM + kb + lk8;
    a.h[0] = *(const v8h*)xr;
    a.h[1] = *(const v8h*)(xr + 16);
    b.h[0] = *(const v8h*)(Bt + kb + lk8);
    b.h[1] = *(const v8h*)(Bt + kb + lk8 + 16);
    acc = wmma_f16(a.v, b.v, acc);
  }
#pragma unroll
  for (int e = 0; e < 8; ++e) red[wave][e * 32 + lane] = acc[e];
  __syncthreads();
  if (wave == 0) {
#pragma unroll
    for (int e = 0; e < 8; ++e) {
      float s = red[0][e * 32 + lane] + red[1][e * 32 + lane] +
                red[2][e * 32 + lane] + red[3][e * 32 + lane];
      int r = row0 + e + ((lane >> 4) << 3);
      h[(size_t)r * RANK + lm] = (_Float16)(2.0f * s);
    }
  }
}

// ------------- Kernel 2: out = x @ M + h_pad @ [B_g; 0] --------------------
// 128x128 tile / 256 threads (8 waves, 4Mx2N). Double-buffered LDS, one
// barrier per K-step. A tile: global_load_async_to_lds_b128 (ASYNCcnt).
// B tile: register load + transposed LDS store (k-contiguous for B-frags).
__global__ __launch_bounds__(256) void lora_gemm_kernel(
    const _Float16* __restrict__ x, const _Float16* __restrict__ hbuf,
    const _Float16* __restrict__ loraB, const _Float16* __restrict__ W,
    _Float16* __restrict__ out) {
  __shared__ _Float16 As[2][BM * LDA];      // row-major rows x k
  __shared__ _Float16 Bs[2][BN * LDA];      // transposed: n-major, k contiguous
  const int tid  = threadIdx.x;
  const int wave = tid >> 5, lane = tid & 31;
  const int wm = wave & 3, wn = wave >> 2;
  const int lm   = lane & 15;
  const int lk8  = (lane >> 4) << 3;
  const int lk16 = (lane >> 4) << 4;
  const int row0 = blockIdx.y * BM;
  const int col0 = blockIdx.x * BN;

  // A tile: thread -> (row, 16-half chunk); 32B contiguous per thread
  const int arow = tid >> 1;
  const int acol = (tid & 1) << 4;
  const _Float16* aG = x + (size_t)(row0 + arow) * FDIM + acol;
  unsigned aLds[2];
  aLds[0] = (unsigned)(uintptr_t)&As[0][arow * LDA + acol];
  aLds[1] = (unsigned)(uintptr_t)&As[1][arow * LDA + acol];

  // B tile: thread -> (k-row pair, 8-col chunk)
  const int bkr = (tid >> 4) << 1;
  const int bc  = (tid & 15) << 3;
  const _Float16* bPtr = W + (size_t)bkr * FDIM + col0 + bc;
  Pack16 pb0, pb1;

  auto asyncA = [&](int k0, int buf) {
    unsigned long long gp = (unsigned long long)(uintptr_t)(aG + k0);
    asm volatile(
      "global_load_async_to_lds_b128 %0, %1, off\n\t"
      "global_load_async_to_lds_b128 %0, %1, off offset:16"
      :: "v"(aLds[buf]), "v"(gp) : "memory");
  };
  auto loadB = [&](int k0) {
    const _Float16* bp = bPtr + (size_t)k0 * FDIM;
    pb0.u = *(const u32x4*)bp;
    pb1.u = *(const u32x4*)(bp + FDIM);
  };
  auto storeB = [&](int buf) {
#pragma unroll
    for (int j = 0; j < 8; ++j) {           // pack (k,k+1) per output col
      h2 p; p.x = pb0.f[j]; p.y = pb1.f[j];
      *(h2*)&Bs[buf][(bc + j) * LDA + bkr] = p;
    }
  };

  v8f acc[2][4];
  const v8f zf = {0.f, 0.f, 0.f, 0.f, 0.f, 0.f, 0.f, 0.f};
#pragma unroll
  for (int mi = 0; mi < 2; ++mi)
#pragma unroll
    for (int ni = 0; ni < 4; ++ni) acc[mi][ni] = zf;

  auto compute = [&](int buf) {
    Frag af[2], bf[4];
#pragma unroll
    for (int mi = 0; mi < 2; ++mi) {
      const _Float16* p = &As[buf][(wm * 32 + mi * 16 + lm) * LDA + lk8];
      af[mi].h[0] = *(const v8h*)p;
      af[mi].h[1] = *(const v8h*)(p + 16);
    }
#pragma unroll
    for (int ni = 0; ni < 4; ++ni) {
      const _Float16* p = &Bs[buf][(wn * 64 + ni * 16 + lm) * LDA + lk8];
      bf[ni].h[0] = *(const v8h*)p;
      bf[ni].h[1] = *(const v8h*)(p + 16);
    }
#pragma unroll
    for (int mi = 0; mi < 2; ++mi)
#pragma unroll
      for (int ni = 0; ni < 4; ++ni)
        acc[mi][ni] = wmma_f16(af[mi].v, bf[ni].v, acc[mi][ni]);
  };

  // prologue: fill buffer 0
  asyncA(0, 0);
  loadB(0);
  storeB(0);
  asm volatile("s_wait_asynccnt 0x0" ::: "memory");
  __syncthreads();

  const int NKT = FDIM / BK;                // 128
  for (int kt = 0; kt < NKT; ++kt) {
    const int cur = kt & 1, nxt = cur ^ 1;
    if (kt + 1 < NKT) {
      asyncA((kt + 1) * BK, nxt);           // async fill of other buffer
      loadB((kt + 1) * BK);
    }
    compute(cur);
    if (kt + 1 < NKT) storeB(nxt);
    asm volatile("s_wait_asynccnt 0x0" ::: "memory");
    __syncthreads();                        // single barrier per K-step
  }

  // ---- LoRA epilogue: one extra K=16 (zero-padded to 32) WMMA step ----
  const int g = row0 / SEG;
  Frag ah[2];
#pragma unroll
  for (int mi = 0; mi < 2; ++mi) {
    int r = row0 + wm * 32 + mi * 16 + lm;
    ah[mi].h[0] = *(const v8h*)&hbuf[(size_t)r * RANK + lk8];   // K 0..15
#pragma unroll
    for (int i = 0; i < 8; ++i) ah[mi].v[8 + i] = (_Float16)0.0f;
  }
  const _Float16* Bg = loraB + (size_t)g * RANK * FDIM;
#pragma unroll
  for (int ni = 0; ni < 4; ++ni) {
    const int cb = col0 + wn * 64 + ni * 16 + lm;
    Frag bh;
#pragma unroll
    for (int i = 0; i < 16; ++i)            // elements 8..15 carry K>=16 -> 0
      bh.v[i] = (i < 8) ? Bg[(size_t)(lk8 + i) * FDIM + cb] : (_Float16)0.0f;
#pragma unroll
    for (int mi = 0; mi < 2; ++mi)
      acc[mi][ni] = wmma_f16(ah[mi].v, bh.v, acc[mi][ni]);
  }

  // ---- store f16 output ----
#pragma unroll
  for (int mi = 0; mi < 2; ++mi)
#pragma unroll
    for (int ni = 0; ni < 4; ++ni) {
      const int c     = col0 + wn * 64 + ni * 16 + lm;
      const int rbase = row0 + wm * 32 + mi * 16 + ((lane >> 4) << 3);
#pragma unroll
      for (int e = 0; e < 8; ++e)
        out[(size_t)(rbase + e) * FDIM + c] = (_Float16)acc[mi][ni][e];
    }
}

extern "C" void kernel_launch(void* const* d_in, const int* in_sizes, int n_in,
                              void* d_out, int out_size, void* d_ws, size_t ws_size,
                              hipStream_t stream) {
  (void)in_sizes; (void)n_in; (void)out_size; (void)ws_size;
  const _Float16* x     = (const _Float16*)d_in[0];   // (4096,1,4096) f16
  const _Float16* loraA = (const _Float16*)d_in[1];   // (8,4096,16)   f16
  const _Float16* loraB = (const _Float16*)d_in[2];   // (8,16,4096)   f16
  const _Float16* M     = (const _Float16*)d_in[3];   // (4096,4096)   f16
  _Float16* out = (_Float16*)d_out;
  _Float16* h   = (_Float16*)d_ws;                    // 4096*16  = 128KB
  _Float16* AgT = h + (size_t)FDIM * RANK;            // 8*16*4096 = 1MB

  transposeA_kernel<<<128, 256, 0, stream>>>(loraA, AgT);
  lora_h_kernel<<<FDIM / 16, 128, 0, stream>>>(x, AgT, h);
  dim3 grid(FDIM / BN, FDIM / BM);
  lora_gemm_kernel<<<grid, 256, 0, stream>>>(x, h, loraB, M, out);
}
